// OpenVocRelCls_LearnablePrompt_9259949490317
// MI455X (gfx1250) — matrix operand order repaired and checked by reading, not compiled, not run on hardware
//
#include <hip/hip_runtime.h>
#include <hip/hip_bf16.h>
#include <stdint.h>

typedef __bf16 bf16;
typedef uint32_t u32;
typedef __attribute__((ext_vector_type(16))) __bf16 v16bf;
typedef __attribute__((ext_vector_type(8)))  __bf16 v8bf;
typedef __attribute__((ext_vector_type(8)))  float   v8f;

#define NUM_BASE  92
#define NUM_NOVEL 40
#define N_LAB     (1 + NUM_BASE + NUM_NOVEL)   /* 133 */
#define ALPHA     0.25f
#define BSZ       16
#define N_DET     48
#define DIM_FEAT  2048
#define DIM_HID   1024
#define DIM_EMB   512
#define EMB_HALF  256
#define PAIRS     (N_DET * (N_DET - 1))        /* 2256  */
#define M_ROWS    (BSZ * PAIRS)                /* 36096 */

/* GEMM tiling: 128x128 block tile, 8 wave32 waves, each wave 64x32 (4x2 WMMA tiles), K-step 32 */
#define BM 128
#define BN 128
#define BK 32
#define LDSK (BK + 8)   /* 40 halfwords = 80 B row stride, 16B-aligned */

#define SHUF16(lo, hi) __builtin_shufflevector((lo), (hi), 0, 1, 2, 3, 4, 5, 6, 7, \
                                               8, 9, 10, 11, 12, 13, 14, 15)

/* ------------------------------------------------------------------ utils */

__global__ void cvt_f32_bf16_kernel(const float* __restrict__ src,
                                    bf16* __restrict__ dst, int n) {
    int i = blockIdx.x * 256 + threadIdx.x;
    if (i < n) dst[i] = (bf16)src[i];
}

__global__ void zero_kernel(float* __restrict__ p, int n) {
    int i = blockIdx.x * 256 + threadIdx.x;
    if (i < n) p[i] = 0.0f;
}

__device__ __forceinline__ float focal_term(float L, float t) {
    /* sigmoid focal loss, gamma=2 */
    float p  = 1.0f / (1.0f + expf(-L));
    float ce = fmaxf(L, 0.0f) - L * t + log1pf(expf(-fabsf(L)));
    float pt = p * t + (1.0f - p) * (1.0f - t);
    float om = 1.0f - pt;
    float at = ALPHA * t + (1.0f - ALPHA) * (1.0f - t);
    return at * ce * om * om;
}

__device__ __forceinline__ float block_reduce_sum(float v, float* red, int tid) {
    red[tid] = v;
    __syncthreads();
    for (int st = 128; st > 0; st >>= 1) {
        if (tid < st) red[tid] += red[tid + st];
        __syncthreads();
    }
    float r = red[0];
    __syncthreads();
    return r;
}

/* --------------------------------------------- GEMM1: gathered A, bf16 WMMA
 * C(M x 1024) = relu( gather(det_feats)(M x 4096) @ W1b + b1 ), stored bf16.
 * K loop split into s-half [0,2048) and o-half [2048,4096).  Each thread's 8
 * gather rows are tracked as running u32 element offsets in registers and the
 * base GEP is taken directly off the global kernel arg, so loads stay on the
 * GLOBAL path (saddr + 32-bit voffset) with one 32-bit add per row per step. */
__global__ __launch_bounds__(256)
void gemm1_gather_kernel(const float* __restrict__ det,
                         const int*   __restrict__ sids,
                         const int*   __restrict__ oids,
                         const bf16*  __restrict__ Wb,    /* (4096,1024) */
                         const float* __restrict__ bias,  /* (1024,)     */
                         bf16*        __restrict__ out)   /* (M,1024)    */
{
    __shared__ bf16 As[BM][LDSK];
    __shared__ bf16 BsT[BN][LDSK];   /* B tile stored transposed: BsT[n][k] */
    __shared__ int  sOffSh[BM];
    __shared__ int  oOffSh[BM];

    const int tid    = threadIdx.x;
    const int lane   = tid & 31;
    const int wave   = tid >> 5;
    const int laneHi = (lane >> 4) & 1;
    const int lm     = lane & 15;

    const int rowBase = blockIdx.y * BM;
    const int colBase = blockIdx.x * BN;

    if (tid < BM) {
        int r  = rowBase + tid;
        int b  = r / PAIRS;
        int pr = r - b * PAIRS;
        sOffSh[tid] = (b * N_DET + sids[pr]) * DIM_FEAT;
        oOffSh[tid] = (b * N_DET + oids[pr]) * DIM_FEAT;
    }
    __syncthreads();

    /* A staging assignment: pair p = i*256+tid -> row = i*16 + (tid>>4),
     * colpair = (tid&15)*2.  8 fixed rows/thread -> 8 running u32 offsets. */
    const int rsub = tid >> 4;
    const int cp   = (tid & 15) * 2;
    u32 offS[8], offO[8];
#pragma unroll
    for (int i = 0; i < 8; ++i) {
        offS[i] = (u32)(sOffSh[i * 16 + rsub] + cp);
        offO[i] = (u32)(oOffSh[i * 16 + rsub] + cp);
    }

    const int waveM = (wave & 1) * 64;
    const int waveN = (wave >> 1) * 32;

    v8f acc[4][2];
#pragma unroll
    for (int i = 0; i < 4; ++i)
#pragma unroll
        for (int j = 0; j < 2; ++j) { v8f z = {}; acc[i][j] = z; }

    /* stage A: gather f32 pairs, packed cvt -> one b32 LDS store; offsets
     * advance by BK elements per step (strength-reduced addressing). */
    auto stageA = [&](u32* offR) {
#pragma unroll
        for (int i = 0; i < 8; ++i) {
            int row = i * 16 + rsub;
            const float2 f = *(const float2*)(det + (size_t)offR[i]);
            offR[i] += BK;
            union { u32 u; bf16 h[2]; } pk;
            pk.h[0] = (bf16)f.x;
            pk.h[1] = (bf16)f.y;
            *(u32*)&As[row][cp] = pk.u;
        }
    };

    /* stage B transposed: unit u = i*256+tid covers 4 cols x 2 k-rows.
     * uint2 loads of two consecutive K rows, halfword-merge, 4x b32 stores. */
    auto stageB = [&](int kBase) {
#pragma unroll
        for (int i = 0; i < 2; ++i) {
            int kk = i * 8 + (tid >> 5);       /* 0..15 -> k = 2*kk */
            int k  = 2 * kk;
            int cc = (tid & 31) * 4;
            const uint2 r0 = *(const uint2*)&Wb[(size_t)(kBase + k)     * DIM_HID + colBase + cc];
            const uint2 r1 = *(const uint2*)&Wb[(size_t)(kBase + k + 1) * DIM_HID + colBase + cc];
            u32 p0 = (r0.x & 0xffffu) | (r1.x << 16);
            u32 p1 = (r0.x >> 16)     | (r1.x & 0xffff0000u);
            u32 p2 = (r0.y & 0xffffu) | (r1.y << 16);
            u32 p3 = (r0.y >> 16)     | (r1.y & 0xffff0000u);
            *(u32*)&BsT[cc + 0][k] = p0;
            *(u32*)&BsT[cc + 1][k] = p1;
            *(u32*)&BsT[cc + 2][k] = p2;
            *(u32*)&BsT[cc + 3][k] = p3;
        }
    };

    /* fragment loads: contiguous 16B runs -> ds_load_b128 pairs */
    auto compute = [&]() {
        v16bf a[4], bfr[2];
#pragma unroll
        for (int mt = 0; mt < 4; ++mt) {
            int  row = waveM + mt * 16 + lm;
            v8bf lo  = *(const v8bf*)&As[row][laneHi * 8];
            v8bf hi  = *(const v8bf*)&As[row][laneHi * 8 + 16];
            a[mt] = SHUF16(lo, hi);
        }
#pragma unroll
        for (int nt = 0; nt < 2; ++nt) {
            int  col = waveN + nt * 16 + lm;
            v8bf lo  = *(const v8bf*)&BsT[col][laneHi * 8];
            v8bf hi  = *(const v8bf*)&BsT[col][laneHi * 8 + 16];
            bfr[nt] = SHUF16(lo, hi);
        }
#pragma unroll
        for (int mt = 0; mt < 4; ++mt)
#pragma unroll
            for (int nt = 0; nt < 2; ++nt)
                acc[mt][nt] = __builtin_amdgcn_wmma_f32_16x16x32_bf16(
                    false, a[mt], false, bfr[nt], (short)0, acc[mt][nt],
                    false, false);
    };

    for (int k0 = 0; k0 < DIM_FEAT; k0 += BK) {      /* subject half */
        stageA(offS);
        stageB(k0);
        __syncthreads();
        compute();
        __syncthreads();
    }
    for (int k0 = 0; k0 < DIM_FEAT; k0 += BK) {      /* object half */
        stageA(offO);
        stageB(k0 + DIM_FEAT);
        __syncthreads();
        compute();
        __syncthreads();
    }

    /* epilogue: bias + relu -> bf16 */
#pragma unroll
    for (int mt = 0; mt < 4; ++mt)
#pragma unroll
        for (int nt = 0; nt < 2; ++nt) {
            int   col = colBase + waveN + nt * 16 + lm;
            float bv  = bias[col];
#pragma unroll
            for (int i = 0; i < 8; ++i) {
                int   row = rowBase + waveM + mt * 16 + i + laneHi * 8;
                float v   = acc[mt][nt][i] + bv;
                v = v > 0.0f ? v : 0.0f;
                out[(size_t)row * DIM_HID + col] = (bf16)v;
            }
        }
}

/* ------------------------------------- generic bf16 WMMA GEMM (GEMM2/GEMM3)
 * C(M x N) = [relu]( A(M x K) @ B(K x N) [+ bias] ); output bf16 or f32. */
template <int N, bool RELU, bool BF16OUT, bool HASBIAS>
__global__ __launch_bounds__(256)
void gemm_bf16_kernel(const bf16*  __restrict__ A, int K,
                      const bf16*  __restrict__ B,
                      const float* __restrict__ bias,
                      bf16*        __restrict__ outB,
                      float*       __restrict__ outF)
{
    __shared__ bf16 As[BM][LDSK];
    __shared__ bf16 BsT[BN][LDSK];

    const int tid    = threadIdx.x;
    const int lane   = tid & 31;
    const int wave   = tid >> 5;
    const int laneHi = (lane >> 4) & 1;
    const int lm     = lane & 15;

    const int rowBase = blockIdx.y * BM;
    const int colBase = blockIdx.x * BN;

    const int waveM = (wave & 1) * 64;
    const int waveN = (wave >> 1) * 32;

    v8f acc[4][2];
#pragma unroll
    for (int i = 0; i < 4; ++i)
#pragma unroll
        for (int j = 0; j < 2; ++j) { v8f z = {}; acc[i][j] = z; }

    /* A staging: 16B vector copies (u = i*256+tid -> row = i*64 + tid>>2) */
    auto stageA = [&](int k0) {
#pragma unroll
        for (int i = 0; i < 2; ++i) {
            int  row  = i * 64 + (tid >> 2);
            int  col8 = (tid & 3) * 8;
            uint4 v = *(const uint4*)&A[(size_t)(rowBase + row) * K + k0 + col8];
            *(uint4*)&As[row][col8] = v;
        }
    };

    auto stageB = [&](int k0) {
#pragma unroll
        for (int i = 0; i < 2; ++i) {
            int kk = i * 8 + (tid >> 5);
            int k  = 2 * kk;
            int cc = (tid & 31) * 4;
            const uint2 r0 = *(const uint2*)&B[(size_t)(k0 + k)     * N + colBase + cc];
            const uint2 r1 = *(const uint2*)&B[(size_t)(k0 + k + 1) * N + colBase + cc];
            u32 p0 = (r0.x & 0xffffu) | (r1.x << 16);
            u32 p1 = (r0.x >> 16)     | (r1.x & 0xffff0000u);
            u32 p2 = (r0.y & 0xffffu) | (r1.y << 16);
            u32 p3 = (r0.y >> 16)     | (r1.y & 0xffff0000u);
            *(u32*)&BsT[cc + 0][k] = p0;
            *(u32*)&BsT[cc + 1][k] = p1;
            *(u32*)&BsT[cc + 2][k] = p2;
            *(u32*)&BsT[cc + 3][k] = p3;
        }
    };

    for (int k0 = 0; k0 < K; k0 += BK) {
        stageA(k0);
        stageB(k0);
        __syncthreads();

        v16bf a[4], bfr[2];
#pragma unroll
        for (int mt = 0; mt < 4; ++mt) {
            int  row = waveM + mt * 16 + lm;
            v8bf lo  = *(const v8bf*)&As[row][laneHi * 8];
            v8bf hi  = *(const v8bf*)&As[row][laneHi * 8 + 16];
            a[mt] = SHUF16(lo, hi);
        }
#pragma unroll
        for (int nt = 0; nt < 2; ++nt) {
            int  col = waveN + nt * 16 + lm;
            v8bf lo  = *(const v8bf*)&BsT[col][laneHi * 8];
            v8bf hi  = *(const v8bf*)&BsT[col][laneHi * 8 + 16];
            bfr[nt] = SHUF16(lo, hi);
        }
#pragma unroll
        for (int mt = 0; mt < 4; ++mt)
#pragma unroll
            for (int nt = 0; nt < 2; ++nt)
                acc[mt][nt] = __builtin_amdgcn_wmma_f32_16x16x32_bf16(
                    false, a[mt], false, bfr[nt], (short)0, acc[mt][nt],
                    false, false);
        __syncthreads();
    }

#pragma unroll
    for (int mt = 0; mt < 4; ++mt)
#pragma unroll
        for (int nt = 0; nt < 2; ++nt) {
            int   col = colBase + waveN + nt * 16 + lm;
            float bv  = HASBIAS ? bias[col] : 0.0f;
#pragma unroll
            for (int i = 0; i < 8; ++i) {
                int   row = rowBase + waveM + mt * 16 + i + laneHi * 8;
                float v   = acc[mt][nt][i] + bv;
                if (RELU) v = v > 0.0f ? v : 0.0f;
                if (BF16OUT) outB[(size_t)row * N + col] = (bf16)v;
                else         outF[(size_t)row * N + col] = v;
            }
        }
}

/* ---------------------------------------------- per-row tail: relpos MLP,
 * norms, logits, focal losses, distillation L1; atomic partial sums. */
__global__ __launch_bounds__(256)
void finalize_rows_kernel(const float* __restrict__ soproj,   /* (M,512)   */
                          const float* __restrict__ relpos,   /* (M,12)    */
                          const float* __restrict__ R1,       /* (12,256)  */
                          const float* __restrict__ rb1,      /* (256,)    */
                          const float* __restrict__ R2,       /* (256,512) */
                          const float* __restrict__ cls,      /* (92,512)  */
                          const float* __restrict__ temperature,
                          const int*   __restrict__ labels,   /* (M,133)   */
                          const float* __restrict__ traj,     /* (B,48,256)*/
                          const int*   __restrict__ sids,
                          const int*   __restrict__ oids,
                          float*       __restrict__ accum)    /* 5 floats  */
{
    __shared__ float t256[256];
    __shared__ float comb[512];
    __shared__ float red[256];

    const int tid = threadIdx.x;
    const int r   = blockIdx.x;
    const int b   = r / PAIRS;
    const int pr  = r - b * PAIRS;
    const int sid = sids[pr];
    const int oid = oids[pr];

    /* relpos hidden: t[j] = relu(rp @ R1 + rb1) */
    const float* rp = relpos + (size_t)r * 12;
    float s = rb1[tid];
#pragma unroll
    for (int k = 0; k < 12; ++k) s += rp[k] * R1[k * 256 + tid];
    t256[tid] = s > 0.0f ? s : 0.0f;
    __syncthreads();

    /* re = t @ R2; each thread owns cols tid and tid+256 */
    float re0 = 0.0f, re1 = 0.0f;
    for (int k = 0; k < 256; ++k) {
        float tk = t256[k];
        re0 += tk * R2[(size_t)k * DIM_EMB + tid];
        re1 += tk * R2[(size_t)k * DIM_EMB + tid + 256];
    }
    float sp0 = soproj[(size_t)r * DIM_EMB + tid];
    float sp1 = soproj[(size_t)r * DIM_EMB + tid + 256];

    float reN = sqrtf(block_reduce_sum(re0 * re0 + re1 * re1, red, tid));
    float spN = sqrtf(block_reduce_sum(sp0 * sp0 + sp1 * sp1, red, tid));
    float invRe = 1.0f / fmaxf(reN, 1e-12f);
    float invSp = 1.0f / fmaxf(spN, 1e-12f);

    float c0 = sp0 * invSp + re0 * invRe;
    float c1 = sp1 * invSp + re1 * invRe;
    float cN = sqrtf(block_reduce_sum(c0 * c0 + c1 * c1, red, tid));
    float invC = 1.0f / fmaxf(cN, 1e-12f);
    comb[tid]       = c0 * invC;
    comb[tid + 256] = c1 * invC;
    __syncthreads();

    /* masks */
    const int* lab = labels + (size_t)r * N_LAB;
    bool neg_mask = lab[0] > 0;
    float pmv = (tid < NUM_BASE && lab[1 + tid] > 0) ? 1.0f : 0.0f;
    bool pos_mask = block_reduce_sum(pmv, red, tid) > 0.0f;

    /* logits + focal terms: thread i < 92 owns class i */
    float invT = 1.0f / temperature[0];
    float posTerm = 0.0f, negTerm = 0.0f;
    if (tid < NUM_BASE) {
        const float* wrow = cls + (size_t)tid * DIM_EMB;
        float L = 0.0f;
        for (int k = 0; k < DIM_EMB; ++k) L += comb[k] * wrow[k];
        L *= invT;
        posTerm = focal_term(L, (float)lab[1 + tid]);
        negTerm = focal_term(L, 0.0f);
    }
    float rowPos = block_reduce_sum(posTerm, red, tid);
    float rowNeg = block_reduce_sum(negTerm, red, tid);

    /* distillation L1: target = concat(traj[sid], traj[oid]) */
    float tg0 = traj[((size_t)b * N_DET + sid) * EMB_HALF + tid];
    float tg1 = traj[((size_t)b * N_DET + oid) * EMB_HALF + tid];
    float rowD = block_reduce_sum(fabsf(tg0 - sp0) + fabsf(tg1 - sp1), red, tid);

    if (tid == 0) {
        if (pos_mask) atomicAdd(&accum[0], rowPos);
        if (neg_mask) atomicAdd(&accum[1], rowNeg);
        atomicAdd(&accum[2], rowD);
        atomicAdd(&accum[3], pos_mask ? 1.0f : 0.0f);
        atomicAdd(&accum[4], neg_mask ? 1.0f : 0.0f);
    }
}

__global__ void final_reduce_kernel(const float* __restrict__ accum,
                                    float* __restrict__ out) {
    float pden = fmaxf(accum[3], 1.0f) * (float)NUM_BASE;
    float nden = fmaxf(accum[4], 1.0f) * (float)NUM_BASE;
    float dmean = accum[2] / ((float)M_ROWS * (float)DIM_EMB);
    out[0] = accum[0] / pden + accum[1] / nden + dmean;
}

/* ------------------------------------------------------------------ launch */

extern "C" void kernel_launch(void* const* d_in, const int* in_sizes, int n_in,
                              void* d_out, int out_size, void* d_ws, size_t ws_size,
                              hipStream_t stream) {
    (void)in_sizes; (void)n_in; (void)out_size; (void)ws_size;

    const float* det    = (const float*)d_in[0];
    const float* traj   = (const float*)d_in[1];
    const float* relpos = (const float*)d_in[2];
    const float* cls    = (const float*)d_in[3];
    const float* W1     = (const float*)d_in[4];
    const float* b1     = (const float*)d_in[5];
    const float* W2     = (const float*)d_in[6];
    const float* b2     = (const float*)d_in[7];
    const float* W3     = (const float*)d_in[8];
    const float* R1     = (const float*)d_in[9];
    const float* rb1    = (const float*)d_in[10];
    const float* R2     = (const float*)d_in[11];
    const float* temp   = (const float*)d_in[12];
    const int*   labels = (const int*)d_in[13];
    const int*   sids   = (const int*)d_in[14];
    const int*   oids   = (const int*)d_in[15];

    char*  ws  = (char*)d_ws;
    size_t off = 0;
    float* accum = (float*)(ws + off);            off += 256;
    bf16*  W1b   = (bf16*)(ws + off);             off += (size_t)2 * DIM_FEAT * DIM_HID * sizeof(bf16);
    bf16*  W2b   = (bf16*)(ws + off);             off += (size_t)DIM_HID * DIM_HID * sizeof(bf16);
    bf16*  W3b   = (bf16*)(ws + off);             off += (size_t)DIM_HID * DIM_EMB * sizeof(bf16);
    bf16*  H1    = (bf16*)(ws + off);             off += (size_t)M_ROWS * DIM_HID * sizeof(bf16);
    bf16*  H2    = (bf16*)(ws + off);             off += (size_t)M_ROWS * DIM_HID * sizeof(bf16);
    float* SOP   = (float*)(ws + off);            off += (size_t)M_ROWS * DIM_EMB * sizeof(float);

    /* weight conversion f32 -> bf16 (L2-resident afterwards) */
    int n1 = 2 * DIM_FEAT * DIM_HID;
    int n2 = DIM_HID * DIM_HID;
    int n3 = DIM_HID * DIM_EMB;
    cvt_f32_bf16_kernel<<<(n1 + 255) / 256, 256, 0, stream>>>(W1, W1b, n1);
    cvt_f32_bf16_kernel<<<(n2 + 255) / 256, 256, 0, stream>>>(W2, W2b, n2);
    cvt_f32_bf16_kernel<<<(n3 + 255) / 256, 256, 0, stream>>>(W3, W3b, n3);
    zero_kernel<<<1, 64, 0, stream>>>(accum, 8);

    /* GEMM1: gathered so @ W1 (+b1, relu) -> H1 bf16 */
    gemm1_gather_kernel<<<dim3(DIM_HID / BN, M_ROWS / BM), 256, 0, stream>>>(
        det, sids, oids, W1b, b1, H1);

    /* GEMM2: H1 @ W2 (+b2, relu) -> H2 bf16 */
    gemm_bf16_kernel<DIM_HID, true, true, true>
        <<<dim3(DIM_HID / BN, M_ROWS / BM), 256, 0, stream>>>(
            H1, DIM_HID, W2b, b2, H2, nullptr);

    /* GEMM3: H2 @ W3 -> SOP f32 */
    gemm_bf16_kernel<DIM_EMB, false, false, false>
        <<<dim3(DIM_EMB / BN, M_ROWS / BM), 256, 0, stream>>>(
            H2, DIM_HID, W3b, nullptr, nullptr, SOP);

    /* per-row tail + loss accumulation */
    finalize_rows_kernel<<<M_ROWS, 256, 0, stream>>>(
        SOP, relpos, R1, rb1, R2, cls, temp, labels, traj, sids, oids, accum);

    final_reduce_kernel<<<1, 1, 0, stream>>>(accum, (float*)d_out);
}
